// StDimLocalLocalContrastModel_62981400428716
// MI455X (gfx1250) — matrix operand order repaired and checked by reading, not compiled, not run on hardware
//
#include <hip/hip_runtime.h>
#include <hip/hip_bf16.h>
#include <stdint.h>

// ---------------------------------------------------------------------------
// StDimLocalLocalContrastModel on MI455X (gfx1250, wave32, WMMA bf16)
//   P=64 spatial, B=1024 batch, C=256 chan, HID=512
//   GEMMs: v_wmma_f32_16x16x32_bf16 ; panels staged via async global->LDS.
// ---------------------------------------------------------------------------

typedef __bf16 bf16_t;
typedef __attribute__((ext_vector_type(16))) __bf16 v16bf;
typedef __attribute__((ext_vector_type(8)))  __bf16 v8bf;
typedef __attribute__((ext_vector_type(8)))  float  v8f;
typedef __attribute__((ext_vector_type(4)))  int    v4i;

#define P_SP 64
#define BATCH 1024
#define CH 256
#define HID 512
#define ROWS (P_SP * BATCH)   // 65536

#define AS1 __attribute__((address_space(1)))
#define AS3 __attribute__((address_space(3)))

// ---- CDNA5 async global->LDS copy (ASYNCcnt path, ISA 08_async_tensor) -----
__device__ __forceinline__ void async_copy_b128(void* lds_dst, const void* g_src) {
#if __has_builtin(__builtin_amdgcn_global_load_async_to_lds_b128)
    __builtin_amdgcn_global_load_async_to_lds_b128(
        (AS1 v4i*)(uintptr_t)g_src,
        (AS3 v4i*)(unsigned)(uintptr_t)lds_dst, 0, 0);
#else
    asm volatile("global_load_async_to_lds_b128 %0, %1, off"
                 :: "v"((unsigned)(uintptr_t)lds_dst), "v"(g_src)
                 : "memory");
#endif
}

__device__ __forceinline__ void async_wait0() {
#if __has_builtin(__builtin_amdgcn_s_wait_asynccnt)
    __builtin_amdgcn_s_wait_asynccnt(0);
#else
    asm volatile("s_wait_asynccnt 0x0" ::: "memory");
#endif
}

// ---- WMMA fragment loaders (contiguous k = 16*kh + s, same order for A and B) -----------------

// A: 16x32 bf16 (MxK), row-major source with element stride `ld`.
// lane L: row r=L&15, kh=L>>4; holds K=[16*kh..16*kh+15]
__device__ __forceinline__ v16bf load_a_frag(const bf16_t* base, int ld, int lane) {
    int r  = lane & 15;
    int kh = lane >> 4;
    const bf16_t* p0 = base + (size_t)r * ld + 16 * kh;
    const bf16_t* p1 = base + (size_t)r * ld + 16 * kh + 8;
    v8bf lo = *(const v8bf*)p0;   // 16B load
    v8bf hi = *(const v8bf*)p1;   // 16B load
    return __builtin_shufflevector(lo, hi, 0,1,2,3,4,5,6,7,8,9,10,11,12,13,14,15);
}

// B: 32x16 bf16 (KxN). B[k][n] = W[n][k] for row-major weight W (ld = K stride).
// lane L: col n=L&15, kh=L>>4; holds K = 16*kh .. 16*kh+15 -> one contiguous 32B load.
__device__ __forceinline__ v16bf load_b_frag(const bf16_t* w_rows, int ld, int k0, int lane) {
    int n  = lane & 15;
    int kh = lane >> 4;
    return *(const v16bf*)(w_rows + (size_t)n * ld + k0 + 16 * kh);
}

__device__ __forceinline__ v8f wmma_bf16(v16bf a, v16bf b, v8f c) {
    return __builtin_amdgcn_wmma_f32_16x16x32_bf16(false, a, false, b, (short)0, c, false, false);
}

// ---- Kernel 1: pack [B,C,H,W] f32 -> [P*B, C] bf16 -------------------------
// out idx o = (p*B + b)*C + c ; in idx = (b*C + c)*64 + p  (H*W = 64)
__global__ __launch_bounds__(256)
void pack_kernel(const float* __restrict__ src, bf16_t* __restrict__ dst) {
    int o = blockIdx.x * 256 + threadIdx.x;          // 0 .. 16777215
    int p   = o >> 18;                               // / (B*C) = / 262144
    int rbc = o & 0x3FFFF;                           // b*C + c
    dst[o] = (bf16_t)src[(size_t)rbc * 64 + p];
}

// ---- Kernel 2: weights f32 -> bf16 ----------------------------------------
__global__ __launch_bounds__(256)
void cvtw_kernel(const float* __restrict__ W1, const float* __restrict__ W2,
                 const float* __restrict__ Ww,
                 bf16_t* __restrict__ W1b, bf16_t* __restrict__ W2b,
                 bf16_t* __restrict__ Wwb) {
    int i = blockIdx.x * 256 + threadIdx.x;          // 0 .. 131071
    if (i < HID * CH) W1b[i] = (bf16_t)W1[i];
    if (i < CH * HID) W2b[i] = (bf16_t)W2[i];
    if (i < CH * CH)  Wwb[i] = (bf16_t)Ww[i];
}

// ---- Kernel 3: fused MLP + residual + projection ---------------------------
// Per workgroup (512 thr = 16 waves): 16 rows of X.
//   xt[16x256]    (LDS, async-staged) = X rows
//   hidden[16x512](LDS) = relu(X W1^T + b1)
//   Y[16x256]     (LDS) = X + hidden W2^T + b2
//   pred[16x256] (global) = Y Ww^T
__global__ __launch_bounds__(512)
void mlp_kernel(const bf16_t* __restrict__ Xa,
                const bf16_t* __restrict__ W1b, const float* __restrict__ b1,
                const bf16_t* __restrict__ W2b, const float* __restrict__ b2,
                const bf16_t* __restrict__ Wwb,
                bf16_t* __restrict__ pred) {
    __shared__ bf16_t xt [16 * CH];    //  8 KB
    __shared__ bf16_t hid[16 * HID];   // 16 KB
    __shared__ bf16_t yb [16 * CH];    //  8 KB

    const int lane = threadIdx.x & 31;
    const int wave = threadIdx.x >> 5;       // 0..15
    const int m0   = blockIdx.x * 16;        // row block base
    const int n_l  = lane & 15;
    const int half = lane >> 4;

    // Stage 0: async-stage the 16x256 X panel (8 KB = 512 threads x 16 B)
    {
        const bf16_t* src = Xa + (size_t)m0 * CH;    // contiguous 8 KB
        async_copy_b128(&xt[threadIdx.x * 8], src + threadIdx.x * 8);
        async_wait0();
    }
    __syncthreads();

    // Stage 1: hidden = relu(X W1^T + b1); 32 col-tiles, 2 per wave
#pragma unroll
    for (int t = 0; t < 2; ++t) {
        const int n0 = (wave * 2 + t) * 16;
        v8f acc = {};
#pragma unroll
        for (int k0 = 0; k0 < CH; k0 += 32) {
            v16bf af = load_a_frag(&xt[k0], CH, lane);
            v16bf bf = load_b_frag(W1b + (size_t)n0 * CH, CH, k0, lane);
            acc = wmma_bf16(af, bf, acc);
        }
        const float bias = b1[n0 + n_l];
#pragma unroll
        for (int j = 0; j < 8; ++j) {
            float v = acc[j] + bias;
            v = v > 0.0f ? v : 0.0f;                 // ReLU
            hid[(j + 8 * half) * HID + n0 + n_l] = (bf16_t)v;
        }
    }
    __syncthreads();

    // Stage 2: Y = X + hidden W2^T + b2; 16 col-tiles, 1 per wave, K=512
    {
        const int n0 = wave * 16;
        v8f acc = {};
#pragma unroll
        for (int k0 = 0; k0 < HID; k0 += 32) {
            v16bf af = load_a_frag(&hid[k0], HID, lane);   // ds_load_b128 x2
            v16bf bf = load_b_frag(W2b + (size_t)n0 * HID, HID, k0, lane);
            acc = wmma_bf16(af, bf, acc);
        }
        const float bias = b2[n0 + n_l];
#pragma unroll
        for (int j = 0; j < 8; ++j) {
            const int m = j + 8 * half;
            float x = (float)xt[m * CH + n0 + n_l];        // residual from LDS
            yb[m * CH + n0 + n_l] = (bf16_t)(acc[j] + bias + x);
        }
    }
    __syncthreads();

    // Stage 3: pred = Y Ww^T; 16 col-tiles, 1 per wave, K=256
    {
        const int n0 = wave * 16;
        v8f acc = {};
#pragma unroll
        for (int k0 = 0; k0 < CH; k0 += 32) {
            v16bf af = load_a_frag(&yb[k0], CH, lane);
            v16bf bf = load_b_frag(Wwb + (size_t)n0 * CH, CH, k0, lane);
            acc = wmma_bf16(af, bf, acc);
        }
#pragma unroll
        for (int j = 0; j < 8; ++j) {
            const int m = j + 8 * half;
            pred[(size_t)(m0 + m) * CH + n0 + n_l] = (bf16_t)acc[j];
        }
    }
}

// ---- Kernel 4: logits[p] = pred[p] @ pos[p]^T, fused row-max subtract ------
// One WG (16 waves) per (p, 16-row block): A-panel async-staged to LDS,
// each wave sweeps 4 column tiles (K=256, 8 WMMAs each), full 16x1024 row
// block accumulated in LDS, per-row max via __shfl_xor, single output write.
__global__ __launch_bounds__(512)
void logits_kernel(const bf16_t* __restrict__ pred,
                   const bf16_t* __restrict__ Xp,
                   float* __restrict__ out) {
    __shared__ bf16_t at[16 * CH];       //  8 KB A panel
    __shared__ float  ob[16 * BATCH];    // 64 KB output block

    const int rb   = blockIdx.x;              // row block 0..63
    const int p    = blockIdx.y;              // spatial  0..63
    const int lane = threadIdx.x & 31;
    const int wave = threadIdx.x >> 5;        // 0..15
    const int n_l  = lane & 15;
    const int half = lane >> 4;

    // async-stage 16x256 bf16 A panel (8 KB)
    {
        const bf16_t* asrc = pred + ((size_t)p * BATCH + rb * 16) * CH;
        async_copy_b128(&at[threadIdx.x * 8], asrc + threadIdx.x * 8);
        async_wait0();
    }
    __syncthreads();

    // each wave computes 4 column tiles: tj = wave + 16*q
#pragma unroll
    for (int q = 0; q < 4; ++q) {
        const int tj = wave + 16 * q;
        const bf16_t* Bm = Xp + ((size_t)p * BATCH + tj * 16) * CH;
        v8f acc = {};
#pragma unroll
        for (int k0 = 0; k0 < CH; k0 += 32) {
            if (k0 + 32 < CH)
                __builtin_prefetch(Bm + k0 + 32, 0, 0);   // global_prefetch_b8
            v16bf af = load_a_frag(&at[k0], CH, lane);
            v16bf bf = load_b_frag(Bm, CH, k0, lane);
            acc = wmma_bf16(af, bf, acc);
        }
#pragma unroll
        for (int j = 0; j < 8; ++j)
            ob[(j + 8 * half) * BATCH + tj * 16 + n_l] = acc[j];
    }
    __syncthreads();

    // row-max + subtract + store; wave w owns output row w of the block
    {
        const float* obr = &ob[wave * BATCH];
        float m = -3.402823466e38f;
#pragma unroll
        for (int i = 0; i < 32; ++i)
            m = fmaxf(m, obr[lane + 32 * i]);
#pragma unroll
        for (int s = 16; s > 0; s >>= 1)
            m = fmaxf(m, __shfl_xor(m, s, 32));
        float* orow = out + ((size_t)p * BATCH + rb * 16 + wave) * BATCH;
#pragma unroll
        for (int i = 0; i < 32; ++i)
            orow[lane + 32 * i] = obr[lane + 32 * i] - m;
    }
}

// ---------------------------------------------------------------------------
extern "C" void kernel_launch(void* const* d_in, const int* in_sizes, int n_in,
                              void* d_out, int out_size, void* d_ws, size_t ws_size,
                              hipStream_t stream) {
    const float* anchor   = (const float*)d_in[0];
    const float* positive = (const float*)d_in[1];
    const float* W1 = (const float*)d_in[2];
    const float* b1 = (const float*)d_in[3];
    const float* W2 = (const float*)d_in[4];
    const float* b2 = (const float*)d_in[5];
    const float* Ww = (const float*)d_in[6];
    float* out = (float*)d_out;

    // workspace layout (bf16): Xa 32MB | Xp 32MB | pred 32MB | weights ~0.6MB
    char* ws = (char*)d_ws;
    const size_t SZ = (size_t)ROWS * CH * sizeof(bf16_t);   // 33554432
    bf16_t* Xa   = (bf16_t*)(ws);
    bf16_t* Xp   = (bf16_t*)(ws + SZ);
    bf16_t* pred = (bf16_t*)(ws + 2 * SZ);
    bf16_t* W1b  = (bf16_t*)(ws + 3 * SZ);
    bf16_t* W2b  = W1b + (size_t)HID * CH;
    bf16_t* Wwb  = W2b + (size_t)CH * HID;

    pack_kernel<<<ROWS * CH / 256, 256, 0, stream>>>(anchor,   Xa);
    pack_kernel<<<ROWS * CH / 256, 256, 0, stream>>>(positive, Xp);
    cvtw_kernel<<<(HID * CH) / 256, 256, 0, stream>>>(W1, W2, Ww, W1b, W2b, Wwb);

    mlp_kernel<<<ROWS / 16, 512, 0, stream>>>(Xa, W1b, b1, W2b, b2, Wwb, pred);

    logits_kernel<<<dim3(P_SP, P_SP), 512, 0, stream>>>(pred, Xp, out);
}
